// BertSelfAttention_48284022342065
// MI455X (gfx1250) — matrix operand order repaired and checked by reading, not compiled, not run on hardware
//
#include <hip/hip_runtime.h>
#include <hip/hip_bf16.h>

typedef __attribute__((ext_vector_type(16))) __bf16 v16bf;
typedef __attribute__((ext_vector_type(8)))  float  v8f;

#define NH    16
#define DH    64
#define SEQ   2048
#define DM    1024
#define NBIN  129
#define BINP  132   // padded bin row

static __device__ __forceinline__ unsigned short f2bf(float f) {
    union { float f; unsigned u; } v; v.f = f;
    unsigned r = v.u + 0x7FFFu + ((v.u >> 16) & 1u);
    return (unsigned short)(r >> 16);
}
static __device__ __forceinline__ float bf2f(unsigned short s) {
    union { unsigned u; float f; } v; v.u = ((unsigned)s) << 16;
    return v.f;
}

union Frag16 { v16bf v; unsigned short s[16]; unsigned int u[8]; };

static __device__ __forceinline__ v8f wmma_bf16(const Frag16& a, const Frag16& b, v8f c) {
    return __builtin_amdgcn_wmma_f32_16x16x32_bf16(false, a.v, false, b.v,
                                                   (short)0, c, false, false);
}

// V workspace swizzle: per (b,h), element (s, dd) lives at
//   (((s>>5)*4 + (dd>>4))*32 + ((dd&15) | (s&16)))*16 + (s&15)
// so that the 32x16 B-fragment for K-block kk, N-tile nt is 32B contiguous per lane.
static __device__ __forceinline__ size_t vswz(int s, int dd) {
    return (size_t)((((s >> 5) * 4 + (dd >> 4)) * 32 + ((dd & 15) | (s & 16))) * 16 + (s & 15));
}

// ---------------------------------------------------------------------------
// Kernel A: Y (bf16) = (X @ W + bias) * outScale.  X:[4096,1024] W:[1024,1024].
// 64x64 tile / workgroup, 8 waves: wave (mw=w&3, nw=w>>2) -> rows mw*16, cols nw*32.
// Bs staged fragment-major so B-frags are contiguous 32B per lane.
// ---------------------------------------------------------------------------
__global__ __launch_bounds__(256) void qkv_gemm(const float* __restrict__ X,
                                                const float* __restrict__ W,
                                                const float* __restrict__ bias,
                                                unsigned short* __restrict__ Y,
                                                float outScale, int vSwizzle)
{
    __shared__ __align__(16) unsigned short As[64 * 32];          // row-major 64x32
    __shared__ __align__(32) unsigned short Bs[4 * 32 * 16];      // [nt][lane][e]

    const int tid  = threadIdx.x;
    const int lane = tid & 31;
    const int wave = tid >> 5;
    const int mw   = wave & 3;
    const int nw   = wave >> 2;
    const int n0   = blockIdx.x * 64;
    const int m0   = blockIdx.y * 64;

    const int M  = lane & 15;
    const int hk = (lane >> 4) << 3;   // 0 or 8 (A-frag K-half select)

    v8f acc0 = {}, acc1 = {};

    // staging coordinates
    const int ra = tid >> 2,      ca = (tid & 3) * 8;   // A: 64 rows x 32 cols, 8 elems/thread
    const int rb = tid >> 3,      cb = (tid & 7) * 8;   // B: 32 rows x 64 cols, 8 elems/thread

    for (int kk = 0; kk < DM; kk += 32) {
        if (kk + 32 < DM) {
            __builtin_prefetch(&X[(size_t)(m0 + ra) * DM + kk + 32 + ca], 0, 1);
            __builtin_prefetch(&W[(size_t)(kk + 32 + rb) * DM + n0 + cb], 0, 1);
        }
        // ---- stage A (vector loads, packed 16B LDS store) ----
        {
            float4 f0 = *(const float4*)&X[(size_t)(m0 + ra) * DM + kk + ca];
            float4 f1 = *(const float4*)&X[(size_t)(m0 + ra) * DM + kk + ca + 4];
            uint4 pk;
            pk.x = (unsigned)f2bf(f0.x) | ((unsigned)f2bf(f0.y) << 16);
            pk.y = (unsigned)f2bf(f0.z) | ((unsigned)f2bf(f0.w) << 16);
            pk.z = (unsigned)f2bf(f1.x) | ((unsigned)f2bf(f1.y) << 16);
            pk.w = (unsigned)f2bf(f1.z) | ((unsigned)f2bf(f1.w) << 16);
            *(uint4*)&As[ra * 32 + ca] = pk;
        }
        // ---- stage B swizzled: dst = ((c>>4)*32 + (c&15) + (r&16))*16 + (r&15) ----
        {
            float4 g0 = *(const float4*)&W[(size_t)(kk + rb) * DM + n0 + cb];
            float4 g1 = *(const float4*)&W[(size_t)(kk + rb) * DM + n0 + cb + 4];
            float vals[8] = { g0.x, g0.y, g0.z, g0.w, g1.x, g1.y, g1.z, g1.w };
            #pragma unroll
            for (int m = 0; m < 8; ++m) {
                int c = cb + m;
                Bs[(((c >> 4) * 32) + (c & 15) + (rb & 16)) * 16 + (rb & 15)] = f2bf(vals[m]);
            }
        }
        __syncthreads();

        Frag16 a;
        *(uint4*)&a.s[0] = *(const uint4*)&As[(mw * 16 + M) * 32 + 2 * hk];
        *(uint4*)&a.s[8] = *(const uint4*)&As[(mw * 16 + M) * 32 + 2 * hk + 8];

        Frag16 b0, b1;
        b0.v = *(const v16bf*)&Bs[((nw * 2) * 32 + lane) * 16];
        b1.v = *(const v16bf*)&Bs[((nw * 2 + 1) * 32 + lane) * 16];

        acc0 = wmma_bf16(a, b0, acc0);
        acc1 = wmma_bf16(a, b1, acc1);
        __syncthreads();
    }

    // write out
    const int rhalf = (lane >> 4) << 3;
    #pragma unroll
    for (int t = 0; t < 2; ++t) {
        v8f acc = t ? acc1 : acc0;
        #pragma unroll
        for (int i = 0; i < 8; ++i) {
            int row = mw * 16 + i + rhalf;
            int col = nw * 32 + t * 16 + (lane & 15);
            int m   = m0 + row;
            int ngl = n0 + col;
            int h   = ngl >> 6;
            int dd  = ngl & 63;
            int bb  = m >> 11;
            int s   = m & 2047;
            float val = (acc[i] + bias[ngl]) * outScale;
            size_t headBase = ((size_t)(bb * NH + h)) * SEQ * DH;
            size_t off = vSwizzle ? vswz(s, dd) : ((size_t)s * DH + dd);
            Y[headBase + off] = f2bf(val);
        }
    }
}

// ---------------------------------------------------------------------------
// Kernel C: attention for one (b,h) and one 16-row q tile.
// Two-pass softmax; each wave owns a 256-column K chunk.
// ---------------------------------------------------------------------------
__global__ __launch_bounds__(256) void attn(const unsigned short* __restrict__ Qb,
                                            const unsigned short* __restrict__ Kb,
                                            const unsigned short* __restrict__ Vb,
                                            const float* __restrict__ mask,
                                            const float* __restrict__ embK,
                                            const float* __restrict__ embV,
                                            float* __restrict__ Out)
{
    __shared__ __align__(16) unsigned short qs[16 * 64];
    __shared__ float qe[16 * BINP];
    __shared__ float bins[16 * BINP];
    __shared__ __align__(16) unsigned short pbuf[8][16 * 32];
    __shared__ float ctxred[8][16 * 64];
    __shared__ float wm[8][16], wl[8][16];
    __shared__ float fm[16], finv[16];
    __shared__ float wtl[8][16], wth[8][16];

    const int tid  = threadIdx.x;
    const int lane = tid & 31;
    const int wave = tid >> 5;
    const int q0 = blockIdx.x * 16;
    const int bh = blockIdx.y;
    const int bb = bh >> 4;
    const size_t base = (size_t)bh * SEQ * DH;
    const float* maskRow = mask + (size_t)bb * SEQ;

    // load q tile (bf16, pre-scaled); zero bins
    {
        int row = tid >> 4, col = (tid & 15) * 4;
        *(uint2*)&qs[row * 64 + col] = *(const uint2*)&Qb[base + (size_t)(q0 + row) * DH + col];
    }
    for (int idx = tid; idx < 16 * BINP; idx += 256) bins[idx] = 0.f;
    __syncthreads();

    // qe[row][r] = q[row] . emb_k[r]
    for (int idx = tid; idx < 16 * NBIN; idx += 256) {
        int row = idx / NBIN, r = idx % NBIN;
        float sum = 0.f;
        for (int dcur = 0; dcur < 64; ++dcur)
            sum += bf2f(qs[row * 64 + dcur]) * embK[r * 64 + dcur];
        qe[row * BINP + r] = sum;
    }
    __syncthreads();

    const int M     = lane & 15;
    const int hk    = (lane >> 4) << 3;
    const int rhalf = (lane >> 4) << 3;
    const int ncol  = lane & 15;
    const int koffB = (lane >= 16) ? 16 : 0;

    // q A-fragments (K = dims 0-31 and 32-63), contiguous 16B chunks
    Frag16 aq0, aq1;
    *(uint4*)&aq0.s[0] = *(const uint4*)&qs[M * 64 + 2 * hk];
    *(uint4*)&aq0.s[8] = *(const uint4*)&qs[M * 64 + 2 * hk + 8];
    *(uint4*)&aq1.s[0] = *(const uint4*)&qs[M * 64 + 32 + 2 * hk];
    *(uint4*)&aq1.s[8] = *(const uint4*)&qs[M * 64 + 40 + 2 * hk];

    const int chunk0 = wave * 256;

    // -------------------- pass 1: online row max / sum --------------------
    float rm[8], rl[8];
    #pragma unroll
    for (int i = 0; i < 8; ++i) { rm[i] = -3.0e38f; rl[i] = 0.f; }

    for (int t = 0; t < 16; ++t) {
        int kc = chunk0 + t * 16;
        const unsigned short* kp = Kb + base + (size_t)(kc + ncol) * DH + koffB;
        __builtin_prefetch(kp + 16 * DH, 0, 1);
        Frag16 bk0, bk1;
        bk0.v = *(const v16bf*)kp;          // 32B contiguous
        bk1.v = *(const v16bf*)(kp + 32);
        v8f c = {};
        c = wmma_bf16(aq0, bk0, c);
        c = wmma_bf16(aq1, bk1, c);

        int   kg = kc + ncol;
        float mk = maskRow[kg];
        #pragma unroll
        for (int i = 0; i < 8; ++i) {
            int row = i + rhalf;
            int rel = (q0 + row) - kg;
            rel = rel > 64 ? 64 : (rel < -64 ? -64 : rel);
            float sv = c[i] + qe[row * BINP + rel + 64] + mk;
            float nm = fmaxf(rm[i], sv);
            rl[i] = rl[i] * __expf(rm[i] - nm) + __expf(sv - nm);
            rm[i] = nm;
        }
    }
    #pragma unroll
    for (int off = 1; off < 16; off <<= 1) {
        #pragma unroll
        for (int i = 0; i < 8; ++i) {
            float om = __shfl_xor(rm[i], off, 32);
            float ol = __shfl_xor(rl[i], off, 32);
            float nm = fmaxf(rm[i], om);
            rl[i] = rl[i] * __expf(rm[i] - nm) + ol * __expf(om - nm);
            rm[i] = nm;
        }
    }
    if ((lane & 15) == 0) {
        #pragma unroll
        for (int i = 0; i < 8; ++i) { wm[wave][rhalf + i] = rm[i]; wl[wave][rhalf + i] = rl[i]; }
    }
    __syncthreads();
    if (tid < 16) {
        float m = wm[0][tid], l = wl[0][tid];
        for (int w = 1; w < 8; ++w) {
            float m2 = wm[w][tid], l2 = wl[w][tid];
            float nm = fmaxf(m, m2);
            l = l * __expf(m - nm) + l2 * __expf(m2 - nm);
            m = nm;
        }
        fm[tid] = m;
        finv[tid] = 1.f / l;
    }
    __syncthreads();

    // -------------------- pass 2: probs, P@V, rel-v bins --------------------
    v8f cv[4] = {};
    float thi[8], tlo[8];
    #pragma unroll
    for (int i = 0; i < 8; ++i) { thi[i] = 0.f; tlo[i] = 0.f; }

    for (int stp = 0; stp < 8; ++stp) {
        int kk = chunk0 + stp * 32;
        #pragma unroll
        for (int t = 0; t < 2; ++t) {
            int kc = kk + t * 16;
            const unsigned short* kp = Kb + base + (size_t)(kc + ncol) * DH + koffB;
            Frag16 bk0, bk1;
            bk0.v = *(const v16bf*)kp;
            bk1.v = *(const v16bf*)(kp + 32);
            v8f c = {};
            c = wmma_bf16(aq0, bk0, c);
            c = wmma_bf16(aq1, bk1, c);

            int   kg = kc + ncol;
            float mk = maskRow[kg];
            #pragma unroll
            for (int i = 0; i < 8; ++i) {
                int row = i + rhalf;
                int rel = (q0 + row) - kg;
                int relc = rel > 64 ? 64 : (rel < -64 ? -64 : rel);
                float sv = c[i] + qe[row * BINP + relc + 64] + mk;
                float p = __expf(sv - fm[row]) * finv[row];
                if (rel >= 64)            thi[i] += p;                    // clamped bin 128
                else if (rel <= -64)      tlo[i] += p;                    // clamped bin 0
                else                      bins[row * BINP + rel + 64] = p; // unique writer
                pbuf[wave][row * 32 + t * 16 + ncol] = f2bf(p);
            }
        }
        __syncthreads();   // pbuf cross-lane visibility, keeps waves in step

        Frag16 ap;
        *(uint4*)&ap.s[0] = *(const uint4*)&pbuf[wave][M * 32 + 2 * hk];
        *(uint4*)&ap.s[8] = *(const uint4*)&pbuf[wave][M * 32 + 2 * hk + 8];

        const unsigned short* vp = Vb + base + (size_t)(((kk >> 5) * 4) * 32 + lane) * 16;
        #pragma unroll
        for (int nt = 0; nt < 4; ++nt) {
            Frag16 bv;
            bv.v = *(const v16bf*)(vp + (size_t)nt * 32 * 16);   // 32B contiguous (swizzled layout)
            cv[nt] = wmma_bf16(ap, bv, cv[nt]);
        }
        __syncthreads();
    }

    // per-wave partial ctx -> LDS
    #pragma unroll
    for (int nt = 0; nt < 4; ++nt)
        #pragma unroll
        for (int i = 0; i < 8; ++i)
            ctxred[wave][(i + rhalf) * 64 + nt * 16 + ncol] = cv[nt][i];

    // tail sums across lanes of same row half
    #pragma unroll
    for (int off = 1; off < 16; off <<= 1) {
        #pragma unroll
        for (int i = 0; i < 8; ++i) {
            thi[i] += __shfl_xor(thi[i], off, 32);
            tlo[i] += __shfl_xor(tlo[i], off, 32);
        }
    }
    if ((lane & 15) == 0) {
        #pragma unroll
        for (int i = 0; i < 8; ++i) { wth[wave][rhalf + i] = thi[i]; wtl[wave][rhalf + i] = tlo[i]; }
    }
    __syncthreads();
    if (tid < 16) {
        float hi = 0.f, lo = 0.f;
        for (int w = 0; w < 8; ++w) { hi += wth[w][tid]; lo += wtl[w][tid]; }
        bins[tid * BINP + 128] = hi;
        bins[tid * BINP + 0]   = lo;
    }
    __syncthreads();

    // final: sum wave partials + bins @ emb_v, store [B,S,DM] fp32
    const int h = bh & 15;
    for (int idx = tid; idx < 16 * 64; idx += 256) {
        int row = idx >> 6, dd = idx & 63;
        float sum = 0.f;
        for (int w = 0; w < 8; ++w) sum += ctxred[w][row * 64 + dd];
        float rsum = 0.f;
        for (int r = 0; r < NBIN; ++r) rsum += bins[row * BINP + r] * embV[r * 64 + dd];
        Out[((size_t)bb * SEQ + q0 + row) * DM + h * 64 + dd] = sum + rsum;
    }
}

// ---------------------------------------------------------------------------
extern "C" void kernel_launch(void* const* d_in, const int* in_sizes, int n_in,
                              void* d_out, int out_size, void* d_ws, size_t ws_size,
                              hipStream_t stream) {
    const float* hs   = (const float*)d_in[0];
    const float* mask = (const float*)d_in[1];
    const float* Wq   = (const float*)d_in[2];
    const float* bq   = (const float*)d_in[3];
    const float* Wk   = (const float*)d_in[4];
    const float* bk   = (const float*)d_in[5];
    const float* Wv   = (const float*)d_in[6];
    const float* bv   = (const float*)d_in[7];
    const float* embK = (const float*)d_in[8];
    const float* embV = (const float*)d_in[9];
    float* out = (float*)d_out;

    const size_t perTensor = (size_t)2 * NH * SEQ * DH;   // 4 Mi bf16 elems
    unsigned short* qb = (unsigned short*)d_ws;
    unsigned short* kb = qb + perTensor;
    unsigned short* vb = kb + perTensor;

    dim3 gA(DM / 64, (2 * SEQ) / 64);   // 16 x 64 tiles
    qkv_gemm<<<gA, 256, 0, stream>>>(hs, Wq, bq, qb, 0.125f, 0);  // q pre-scaled 1/sqrt(64)
    qkv_gemm<<<gA, 256, 0, stream>>>(hs, Wk, bk, kb, 1.0f, 0);
    qkv_gemm<<<gA, 256, 0, stream>>>(hs, Wv, bv, vb, 1.0f, 1);    // V fragment-swizzled

    dim3 gC(SEQ / 16, 2 * NH);          // 128 q-tiles x (B*H) heads
    attn<<<gC, 256, 0, stream>>>(qb, kb, vb, mask, embK, embV, out);
}
